// RelevanceMatchingModel_20083267076208
// MI455X (gfx1250) — compile-verified, matrix operand order checked by reading, not run on hardware
//
#include <hip/hip_runtime.h>
#include <hip/hip_bf16.h>

// RelevanceMatchingModel for MI455X / gfx1250.
// prep_kernel: W1,W2 -> bf16 transposed padded (in d_ws).
// rel_kernel: one block per (batch, snippet); 4 GEMM stages on
//   v_wmma_f32_16x16x32_bf16; zero-padded LDS tiles, b128 fragment loads,
//   shared-operand tiling (multiple independent WMMAs per fragment load).
// topk_kernel: top-5 + Wk dot + idf softmax.
// d_ws layout: rel[32][50][100] f32 (640,000 B) | W1t bf16[128][320] (81,920 B)
//              | W2t bf16[64][128] (16,384 B)   -> 738,304 B total.

#define B_    32
#define NP_   2000
#define QL_   50
#define S_    100
#define L_    60
#define E_    300
#define DD_   100
#define DH_   50
#define K_    5
#define NEG_  1.0e12f

#define SMEM_BYTES 108576

typedef __attribute__((ext_vector_type(16))) __bf16 v16bf;
typedef __attribute__((ext_vector_type(8)))  __bf16 v8bf;
typedef __attribute__((ext_vector_type(8)))  float  v8f;

__device__ __forceinline__ unsigned short f2bf(float f) {
  unsigned int u = __builtin_bit_cast(unsigned int, f);
  u += 0x7FFFu + ((u >> 16) & 1u);            // round-to-nearest-even
  return (unsigned short)(u >> 16);
}
__device__ __forceinline__ float bf2f(unsigned short h) {
  unsigned int u = ((unsigned int)h) << 16;
  return __builtin_bit_cast(float, u);
}
__device__ __forceinline__ __bf16 us2bf(unsigned short h) {
  return __builtin_bit_cast(__bf16, h);
}

// Table-order-k 16-bit fragment: elements [0..7] = row[kb + hi*8 + 0..7],
// [8..15] = row[kb + 16 + hi*8 + 0..7] -> two b128 loads + shuffle.
__device__ __forceinline__ v16bf ldfrag(const unsigned short* row, int kb, int laneHi) {
  const v8bf lo = *(const v8bf*)(row + kb + laneHi * 8);
  const v8bf hi = *(const v8bf*)(row + kb + 16 + laneHi * 8);
  return __builtin_shufflevector(lo, hi, 0, 1, 2, 3, 4, 5, 6, 7,
                                 8, 9, 10, 11, 12, 13, 14, 15);
}

__global__ __launch_bounds__(256) void prep_kernel(
    const float* __restrict__ W1, const float* __restrict__ W2,
    unsigned short* __restrict__ W1t, unsigned short* __restrict__ W2t)
{
  int i = blockIdx.x * 256 + threadIdx.x;
  if (i < 128 * 320) {
    int d = i / 320, e = i % 320;
    W1t[i] = (d < DD_ && e < E_) ? f2bf(W1[e * DD_ + d]) : (unsigned short)0;
  } else if (i < 128 * 320 + 64 * 128) {
    int j = i - 128 * 320;
    int d2 = j / 128, dk = j % 128;
    W2t[j] = (d2 < DH_ && dk < DD_) ? f2bf(W2[dk * DH_ + d2]) : (unsigned short)0;
  }
}

__global__ __launch_bounds__(256) void rel_kernel(
    const int* __restrict__ r_idx, const int* __restrict__ c_idx,
    const float* __restrict__ matrix, const float* __restrict__ report_table,
    const float* __restrict__ report_word_emb,
    const float* __restrict__ code_table, const float* __restrict__ snippet_table,
    const float* __restrict__ code_word_emb,
    const float* __restrict__ snippet_len_table,
    const float* __restrict__ code_len_table,
    const unsigned short* __restrict__ W1t, const float* __restrict__ b1,
    const unsigned short* __restrict__ W2t, const float* __restrict__ b2,
    const float* __restrict__ W3, const float* __restrict__ b3,
    float* __restrict__ rel_ws)
{
  extern __shared__ unsigned char smem[];
  unsigned short* c_bf = (unsigned short*)(smem);            // [64][320] bf16; later attn_v
  unsigned short* r_bf = (unsigned short*)(smem + 40960);    // [64][320] bf16; later rc
  float*          sc   = (float*)(smem + 81920);             // scores [64][64] f32
  unsigned short* h1   = (unsigned short*)(smem + 81920);    // overlay: h1 [64][128] bf16
  unsigned short* a_bf = (unsigned short*)(smem + 98304);    // attn [64][64] bf16
  unsigned short* h2   = (unsigned short*)(smem + 98304);    // overlay: h2 [64][64] bf16
  int*   cw    = (int*)(smem + 106496);                      // [64]
  int*   rw    = (int*)(smem + 106752);                      // [64]
  float* rnorm = (float*)(smem + 107008);                    // [64]
  float* vnorm = (float*)(smem + 107264);                    // [64]
  float* b1l   = (float*)(smem + 107520);                    // [128] padded
  float* b2l   = (float*)(smem + 108032);                    // [64] padded
  float* w3l   = (float*)(smem + 108288);                    // [64] padded
  int*   scal  = (int*)(smem + 108544);                      // [4]
  float* fscal = (float*)(smem + 108560);                    // [4]

  const int tid    = threadIdx.x;
  const int wid    = tid >> 5;
  const int lane   = tid & 31;
  const int laneLo = lane & 15;
  const int laneHi = lane >> 4;
  const int b = blockIdx.x / S_;
  const int s = blockIdx.x % S_;

  if (tid == 0) {
    int r = r_idx[b];
    int c = c_idx[b];
    int trans = (int)matrix[(size_t)r * NP_ + c];
    int snip  = (int)code_table[(size_t)trans * S_ + s];
    scal[0] = (int)snippet_len_table[snip];
    scal[1] = (int)code_len_table[c];
    scal[2] = r;
    scal[3] = snip;
    fscal[0] = b3[0];
  }
  __syncthreads();
  const int slen = scal[0];
  const int clen = scal[1];
  const int rrep = scal[2];
  const int snip = scal[3];

  if (tid < 64) {
    int w = (tid < L_) ? (int)snippet_table[(size_t)snip * L_ + tid] : 0;
    cw[tid] = w;
    if (tid < L_) __builtin_prefetch(code_word_emb + (size_t)w * E_, 0, 0);
    rw[tid] = (tid < QL_) ? (int)report_table[(size_t)rrep * QL_ + tid] : 0;
  } else if (tid < 128) {
    int j = tid - 64;
    b2l[j] = (j < DH_) ? b2[j] : 0.f;
    w3l[j] = (j < DH_) ? W3[j] : 0.f;
  } else {
    int j = tid - 128;
    b1l[j] = (j < DD_) ? b1[j] : 0.f;
  }
  __syncthreads();

  // Gather embeddings into zero-padded bf16 LDS tiles (float4 -> 4x bf16).
  for (int i = tid; i < 64 * 80; i += 256) {
    int l = i / 80, e4 = (i % 80) * 4;
    uint2 ov;
    if (l < L_ && e4 < E_) {
      const float4 vv = *(const float4*)(code_word_emb + (size_t)cw[l] * E_ + e4);
      ov.x = (unsigned)f2bf(vv.x) | ((unsigned)f2bf(vv.y) << 16);
      ov.y = (unsigned)f2bf(vv.z) | ((unsigned)f2bf(vv.w) << 16);
    } else { ov.x = 0u; ov.y = 0u; }
    *(uint2*)(c_bf + l * 320 + e4) = ov;
  }
  for (int i = tid; i < 64 * 80; i += 256) {
    int q = i / 80, e4 = (i % 80) * 4;
    uint2 ov;
    if (q < QL_ && e4 < E_) {
      const float4 vv = *(const float4*)(report_word_emb + (size_t)rw[q] * E_ + e4);
      ov.x = (unsigned)f2bf(vv.x) | ((unsigned)f2bf(vv.y) << 16);
      ov.y = (unsigned)f2bf(vv.z) | ((unsigned)f2bf(vv.w) << 16);
    } else { ov.x = 0u; ov.y = 0u; }
    *(uint2*)(r_bf + q * 320 + e4) = ov;
  }
  __syncthreads();

  if (tid < 64) {     // report row sumsq from LDS copy (padding rows -> 0)
    float ss = 0.f;
    for (int e = 0; e < E_; ++e) { float x = bf2f(r_bf[tid * 320 + e]); ss += x * x; }
    rnorm[tid] = ss;
  }
  __syncthreads();

  // ---- GEMM1: scores[q][l] = sum_e r[q][e]*c[l][e] ----
  // Wave w owns tiles (mt0, nt) and (mt0+2, nt): shared B, 2 WMMAs per load.
  {
    const int nt  = wid & 3;
    const int mt0 = wid >> 2;
    const unsigned short* brow  = c_bf + (nt * 16 + laneLo) * 320;
    const unsigned short* arow0 = r_bf + (mt0 * 16 + laneLo) * 320;
    const unsigned short* arow1 = arow0 + 32 * 320;
    v8f acc0 = {}, acc1 = {};
#pragma unroll 1
    for (int kb = 0; kb < 320; kb += 32) {
      v16bf bv  = ldfrag(brow,  kb, laneHi);
      v16bf av0 = ldfrag(arow0, kb, laneHi);
      v16bf av1 = ldfrag(arow1, kb, laneHi);
      acc0 = __builtin_amdgcn_wmma_f32_16x16x32_bf16(false, av0, false, bv,
                                                     (short)0, acc0, false, false);
      acc1 = __builtin_amdgcn_wmma_f32_16x16x32_bf16(false, av1, false, bv,
                                                     (short)0, acc1, false, false);
    }
#pragma unroll
    for (int rr = 0; rr < 8; ++rr) {
      int n = nt * 16 + laneLo;
      sc[(mt0 * 16 + laneHi * 8 + rr) * 64 + n]        = acc0[rr];
      sc[((mt0 + 2) * 16 + laneHi * 8 + rr) * 64 + n]  = acc1[rr];
    }
  }
  __syncthreads();

  // ---- masked softmax over l -> attn probs (rows >= QL_ zeroed) ----
  if (tid < 64) {
    int q = tid;
    if (q < QL_) {
      float mx = -3.4e38f;
      for (int l = 0; l < slen; ++l) mx = fmaxf(mx, sc[q * 64 + l]);
      float sum = 0.f;
      for (int l = 0; l < slen; ++l) sum += __expf(sc[q * 64 + l] - mx);
      float inv = 1.f / sum;
      for (int l = 0; l < 64; ++l) {
        float a = (l < slen) ? __expf(sc[q * 64 + l] - mx) * inv : 0.f;
        a_bf[q * 64 + l] = f2bf(a);
      }
    } else {
      for (int l = 0; l < 64; ++l) a_bf[q * 64 + l] = 0;
    }
  }
  __syncthreads();

  // ---- GEMM2: attn_v[q][e] = sum_l a[q][l]*c[l][e] ----
  // Wave w: fixed m-tile (mt2), 10 n-tiles (ntb+2i): A loaded once per k-step,
  // 10 independent WMMAs per A fragment.
  const int mt2 = wid & 3;
  const int ntb = wid >> 2;
  v8f acc2[10] = {};
  {
    const unsigned short* arow = a_bf + (mt2 * 16 + laneLo) * 64;
#pragma unroll
    for (int kb = 0; kb < 64; kb += 32) {
      v16bf av = ldfrag(arow, kb, laneHi);
#pragma unroll
      for (int i = 0; i < 10; ++i) {
        int e = (ntb + 2 * i) * 16 + laneLo;
        v16bf bv;     // strided read of c_bf as B[k=l][n=e] (unguarded, padded)
#pragma unroll
        for (int v = 0; v < 8; ++v) {
#pragma unroll
          for (int j = 0; j < 2; ++j) {
            int k = (v < 4 ? 2 * v : 16 + 2 * (v - 4)) + (laneHi ? 8 : 0) + j;
            bv[2 * v + j] = us2bf(c_bf[(kb + k) * 320 + e]);
          }
        }
        acc2[i] = __builtin_amdgcn_wmma_f32_16x16x32_bf16(false, av, false, bv,
                                                          (short)0, acc2[i], false, false);
      }
    }
  }
  __syncthreads();     // all c_bf reads done; overlay attn_v into same region
#pragma unroll
  for (int i = 0; i < 10; ++i) {
    int e = (ntb + 2 * i) * 16 + laneLo;
#pragma unroll
    for (int rr = 0; rr < 8; ++rr) {
      int q = mt2 * 16 + laneHi * 8 + rr;
      c_bf[q * 320 + e] = (q < QL_ && e < E_) ? f2bf(acc2[i][rr]) : (unsigned short)0;
    }
  }
  __syncthreads();
  if (tid < 64) {      // attn_v row sumsq (padding rows -> 0)
    float ss = 0.f;
    for (int e = 0; e < E_; ++e) { float x = bf2f(c_bf[tid * 320 + e]); ss += x * x; }
    vnorm[tid] = ss;
  }
  __syncthreads();

  // ---- rc = (r/|r|) .* (v/|v|), in-place over r_bf (padding stays 0) ----
  for (int i = tid; i < 64 * 320; i += 256) {
    int q = i / 320;
    float rn = rnorm[q]; float invr = rn > 0.f ? rsqrtf(rn) : 0.f;
    float vn = vnorm[q]; float invv = vn > 0.f ? rsqrtf(vn) : 0.f;
    r_bf[i] = f2bf(bf2f(r_bf[i]) * invr * bf2f(c_bf[i]) * invv);
  }
  __syncthreads();

  // ---- GEMM3: h1 = tanh(rc @ W1 + b1) ----
  // Wave w: fixed n-tile (B row from global W1t, loaded once per k-step),
  // 4 m-tiles -> 4 independent WMMAs per B fragment.
  {
    const unsigned short* brow = W1t + (size_t)(wid * 16 + laneLo) * 320;
    v8f acc3[4] = {};
#pragma unroll 1
    for (int kb = 0; kb < 320; kb += 32) {
      v16bf bv = ldfrag(brow, kb, laneHi);
#pragma unroll
      for (int it = 0; it < 4; ++it) {
        v16bf av = ldfrag(r_bf + (it * 16 + laneLo) * 320, kb, laneHi);
        acc3[it] = __builtin_amdgcn_wmma_f32_16x16x32_bf16(false, av, false, bv,
                                                           (short)0, acc3[it], false, false);
      }
    }
    int d = wid * 16 + laneLo;
#pragma unroll
    for (int it = 0; it < 4; ++it) {
#pragma unroll
      for (int rr = 0; rr < 8; ++rr) {
        int m = it * 16 + laneHi * 8 + rr;
        h1[m * 128 + d] = (m < QL_ && d < DD_)
                          ? f2bf(tanhf(acc3[it][rr] + b1l[d])) : (unsigned short)0;
      }
    }
  }
  __syncthreads();

  // ---- GEMM4: h2 = tanh(h1 @ W2 + b2) ----
  // Wave w owns tiles (mt0, nt) and (mt0+2, nt): shared B (global W2t).
  {
    const int nt  = wid & 3;
    const int mt0 = wid >> 2;
    const unsigned short* brow  = W2t + (size_t)(nt * 16 + laneLo) * 128;
    const unsigned short* arow0 = h1 + (mt0 * 16 + laneLo) * 128;
    const unsigned short* arow1 = arow0 + 32 * 128;
    v8f acc0 = {}, acc1 = {};
#pragma unroll
    for (int kb = 0; kb < 128; kb += 32) {
      v16bf bv  = ldfrag(brow,  kb, laneHi);
      v16bf av0 = ldfrag(arow0, kb, laneHi);
      v16bf av1 = ldfrag(arow1, kb, laneHi);
      acc0 = __builtin_amdgcn_wmma_f32_16x16x32_bf16(false, av0, false, bv,
                                                     (short)0, acc0, false, false);
      acc1 = __builtin_amdgcn_wmma_f32_16x16x32_bf16(false, av1, false, bv,
                                                     (short)0, acc1, false, false);
    }
    int d2 = nt * 16 + laneLo;
#pragma unroll
    for (int rr = 0; rr < 8; ++rr) {
      int m0 = mt0 * 16 + laneHi * 8 + rr;
      int m1 = m0 + 32;
      h2[m0 * 64 + d2] = (m0 < QL_ && d2 < DH_)
                         ? f2bf(tanhf(acc0[rr] + b2l[d2])) : (unsigned short)0;
      h2[m1 * 64 + d2] = (m1 < QL_ && d2 < DH_)
                         ? f2bf(tanhf(acc1[rr] + b2l[d2])) : (unsigned short)0;
    }
  }
  __syncthreads();

  // ---- layer 3 + snippet mask, store rel[b][q][s] ----
  if (tid < QL_) {
    float acc = fscal[0];
    for (int d = 0; d < DH_; ++d) acc += bf2f(h2[tid * 64 + d]) * w3l[d];
    float rel = tanhf(acc);
    if (s >= clen) rel = 0.f;
    rel_ws[((size_t)b * QL_ + tid) * S_ + s] = rel;
  }
}

__global__ __launch_bounds__(64) void topk_kernel(
    const float* __restrict__ rel_ws, const int* __restrict__ r_idx,
    const float* __restrict__ report_table, const float* __restrict__ idf_table,
    const float* __restrict__ Wk, const float* __restrict__ bk,
    float* __restrict__ out)
{
  __shared__ float s_rc[QL_];
  __shared__ float s_idf[QL_];
  int b = blockIdx.x;
  int t = threadIdx.x;
  if (t < QL_) {
    const float* row = rel_ws + ((size_t)b * QL_ + t) * S_;
    float top[K_];
#pragma unroll
    for (int j = 0; j < K_; ++j) top[j] = -3.4e38f;
    for (int si = 0; si < S_; ++si) {
      float v = row[si];
#pragma unroll
      for (int j = 0; j < K_; ++j) {
        if (v > top[j]) {
#pragma unroll
          for (int m = K_ - 1; m > j; --m) top[m] = top[m - 1];
          top[j] = v;
          break;
        }
      }
    }
    float rc = bk[0];
#pragma unroll
    for (int j = 0; j < K_; ++j) rc += top[j] * Wk[j];
    int r  = r_idx[b];
    int wq = (int)report_table[(size_t)r * QL_ + t];
    float idf = idf_table[wq];
    if (idf == 0.f) idf = -NEG_;
    s_rc[t]  = rc;
    s_idf[t] = idf;
  }
  __syncthreads();
  if (t == 0) {
    float mx = -3.4e38f;
    for (int q = 0; q < QL_; ++q) mx = fmaxf(mx, s_idf[q]);
    float sum = 0.f;
    for (int q = 0; q < QL_; ++q) sum += __expf(s_idf[q] - mx);
    float o = 0.f;
    for (int q = 0; q < QL_; ++q) o += s_rc[q] * (__expf(s_idf[q] - mx) / sum);
    out[b] = o;
  }
}

extern "C" void kernel_launch(void* const* d_in, const int* in_sizes, int n_in,
                              void* d_out, int out_size, void* d_ws, size_t ws_size,
                              hipStream_t stream) {
  (void)in_sizes; (void)n_in; (void)out_size; (void)ws_size;
  const int*   r_idx             = (const int*)d_in[0];
  const int*   c_idx             = (const int*)d_in[1];
  const float* matrix            = (const float*)d_in[2];
  const float* report_table      = (const float*)d_in[3];
  const float* report_word_emb   = (const float*)d_in[4];
  const float* idf_table         = (const float*)d_in[5];
  const float* code_table        = (const float*)d_in[6];
  const float* snippet_table     = (const float*)d_in[7];
  const float* code_word_emb     = (const float*)d_in[8];
  const float* snippet_len_table = (const float*)d_in[9];
  const float* code_len_table    = (const float*)d_in[10];
  const float* W1 = (const float*)d_in[11];
  const float* b1 = (const float*)d_in[12];
  const float* W2 = (const float*)d_in[13];
  const float* b2 = (const float*)d_in[14];
  const float* W3 = (const float*)d_in[15];
  const float* b3 = (const float*)d_in[16];
  const float* Wk = (const float*)d_in[17];
  const float* bk = (const float*)d_in[18];

  float* rel_ws = (float*)d_ws;                                   // 640,000 B
  unsigned short* W1t = (unsigned short*)((char*)d_ws + 640000);  //  81,920 B
  unsigned short* W2t = (unsigned short*)((char*)d_ws + 721920);  //  16,384 B

  prep_kernel<<<192, 256, 0, stream>>>(W1, W2, W1t, W2t);

  rel_kernel<<<B_ * S_, 256, SMEM_BYTES, stream>>>(
      r_idx, c_idx, matrix, report_table, report_word_emb,
      code_table, snippet_table, code_word_emb,
      snippet_len_table, code_len_table,
      W1t, b1, W2t, b2, W3, b3, rel_ws);

  topk_kernel<<<B_, 64, 0, stream>>>(rel_ws, r_idx, report_table, idf_table,
                                     Wk, bk, (float*)d_out);
}
